// SparseAttention_64845416235553
// MI455X (gfx1250) — matrix operand order repaired and checked by reading, not compiled, not run on hardware
//
#include <hip/hip_runtime.h>
#include <hip/hip_bf16.h>
#include <math.h>

// ---------------------------------------------------------------------------
// SparseAttention for MI455X (gfx1250, wave32, WMMA f32_16x16x32_f16)
//   b=1, h=24, n=3072, d=128, QB=192 (m=16 groups), topk=512
// K0: K f32 -> f16 pre-conversion (enables async byte-copy staging)
// K1: colsum attention (S^T stats pass + S colsum pass), double-buffered
//     async K-tile staging via global_load_async_to_lds_b128
// K2: topk mask + compacted column index list (block scan)
// K3: gathered block-sparse flash attention (compacted columns only)
// ---------------------------------------------------------------------------

typedef __attribute__((ext_vector_type(16))) _Float16 v16h;
typedef __attribute__((ext_vector_type(8)))  _Float16 h8v;
typedef __attribute__((ext_vector_type(4)))  _Float16 h4v;
typedef __attribute__((ext_vector_type(8)))  float    v8f;

#define H_   24
#define N_   3072
#define D_   128
#define QB_  192
#define M_   16
#define SCALE_ 0.08838834764831845f   // 1/sqrt(128)
#define NEGBIG_ (-1.0e30f)

__device__ __forceinline__ v8f wmma_f16(v16h a, v16h b, v8f c) {
  return __builtin_amdgcn_wmma_f32_16x16x32_f16(false, a, false, b, (short)0, c,
                                                false, false);
}
__device__ __forceinline__ v16h pack16(h8v lo, h8v hi) {
  return __builtin_shufflevector(lo, hi, 0,1,2,3,4,5,6,7,8,9,10,11,12,13,14,15);
}

// --- CDNA5 async copy: 16B global -> LDS, tracked by ASYNCcnt --------------
// LDS offset = low 32 bits of the generic pointer to shared memory
// (flat LDS aperture).
__device__ __forceinline__ unsigned lds_off32(const void* p) {
  return (unsigned)(unsigned long long)p;
}
__device__ __forceinline__ void async_ld16(const void* gsrc, unsigned ldsOff) {
  asm volatile("global_load_async_to_lds_b128 %0, %1, off"
               :: "v"(ldsOff), "v"(gsrc) : "memory");
}
__device__ __forceinline__ void wait_async0() {
  asm volatile("s_wait_asynccnt 0x0" ::: "memory");
}

// A fragment (16x32 f16, MxK) from global f32 row pointer.
// lane<16 -> K {0..7} u {16..23}; lane>=16 -> K {8..15} u {24..31}
__device__ __forceinline__ v16h load_a_f32(const float* rowp, int k0, int lane) {
  const int kb = ((lane >> 4) << 3);
  const float* p0 = rowp + k0 + kb;
  const float* p1 = rowp + k0 + 16 + kb;
  v16h r;
#pragma unroll
  for (int i = 0; i < 8; ++i) r[i]     = (_Float16)p0[i];
#pragma unroll
  for (int i = 0; i < 8; ++i) r[8 + i] = (_Float16)p1[i];
  return r;
}

// A fragment from LDS f16 tile, row stride 128 halves.
__device__ __forceinline__ v16h load_a_lds(const _Float16* tile, int k0, int lane) {
  const int row = lane & 15;
  const int kb  = ((lane >> 4) << 3);
  const _Float16* p = tile + row * D_ + k0 + kb;
  h8v lo = *(const h8v*)(p);
  h8v hi = *(const h8v*)(p + 16);
  return pack16(lo, hi);
}

// B fragment (32x16 f16, KxN): lane<16 -> col N=lane, K {0..7} u {16..23}; lane>=16 -> K {8..15} u {24..31}.
__device__ __forceinline__ v16h load_b_lds(const _Float16* base, int lane) {
  const int kb = ((lane >> 4) << 3);
  h8v lo = *(const h8v*)(base + kb);
  h8v hi = *(const h8v*)(base + kb + 16);
  return pack16(lo, hi);
}

// B fragment of Q^T from a global f32 Q row pointer (K = d contiguous).
__device__ __forceinline__ v16h load_b_f32(const float* rowp, int k0, int lane) {
  const int kb = ((lane >> 4) << 3);
  const float* p = rowp + k0 + kb;
  v16h r;
#pragma unroll
  for (int i = 0; i < 16; ++i) r[i] = (_Float16)p[(i & 7) + ((i >> 3) << 4)];
  return r;
}

// issue one 32x128 f16 K tile as async 16B chunks (512 chunks over 384 thr)
__device__ __forceinline__ void issue_ktile(const _Float16* __restrict__ Kh,
                                            int hb, int ct, unsigned dstOff,
                                            int tid) {
  for (int i = tid; i < 512; i += 384) {
    const int e = i << 3, r = e >> 7, c = e & 127;
    async_ld16(Kh + ((size_t)(hb * N_ + ct * 32 + r)) * D_ + c,
               dstOff + (unsigned)((r * D_ + c) * 2));
  }
}

// ---------------------------------------------------------------------------
// Kernel 0: K f32 -> f16
// ---------------------------------------------------------------------------
__global__ __launch_bounds__(256) void convert_k_kernel(
    const float* __restrict__ K, _Float16* __restrict__ Kh) {
  const size_t e = ((size_t)blockIdx.x * 256 + threadIdx.x) << 2;
  if (e >= (size_t)H_ * N_ * D_) return;
  const float4 kv = *(const float4*)&K[e];
  h4v hk = {(_Float16)kv.x, (_Float16)kv.y, (_Float16)kv.z, (_Float16)kv.w};
  *(h4v*)&Kh[e] = hk;
}

// ---------------------------------------------------------------------------
// Kernel 1: exact softmax column sums per (h,m) group.
// 384 threads = 12 waves; wave w owns rows [m*192+16w, +16).
// Pass A: S^T = K.Q^T (per-lane row stats + one xor-16 combine).
// Pass B: S = Q.K^T, one ds atomic per subtile into LDS cs[].
// K tiles double-buffered through the async-to-LDS engine.
// ---------------------------------------------------------------------------
__global__ __launch_bounds__(384) void colsum_kernel(
    const float* __restrict__ Q, const _Float16* __restrict__ Kh,
    float* __restrict__ CS) {
  const int hb   = blockIdx.x / M_;
  const int mb   = blockIdx.x % M_;
  const int tid  = threadIdx.x;
  const int wave = tid >> 5;
  const int lane = tid & 31;
  const int lcol = lane & 15;
  const int shl  = (lane >> 4) << 3;

  __shared__ __align__(16) _Float16 sK[2 * 32 * D_];   // double buffer
  __shared__ float sCs[N_];

  const unsigned sKoff[2] = {lds_off32(&sK[0]), lds_off32(&sK[32 * D_])};

  for (int i = tid; i < N_; i += 384) sCs[i] = 0.0f;

  const int row0 = mb * QB_ + wave * 16;
  const float* qrow = Q + ((size_t)(hb * N_ + row0 + lcol)) * D_;

  // ---- Pass A ----
  v16h bQ[4];
#pragma unroll
  for (int f = 0; f < 4; ++f) bQ[f] = load_b_f32(qrow, 32 * f, lane);

  float mrun = -INFINITY, lrun = 0.0f;

  issue_ktile(Kh, hb, 0, sKoff[0], tid);
#pragma unroll 1
  for (int ct = 0; ct < N_ / 32; ++ct) {
    wait_async0();
    __syncthreads();
    if (ct + 1 < N_ / 32) issue_ktile(Kh, hb, ct + 1, sKoff[(ct + 1) & 1], tid);
    const _Float16* cur = sK + (ct & 1) * 32 * D_;

    v16h kf0[4], kf1[4];
#pragma unroll
    for (int f = 0; f < 4; ++f) {
      kf0[f] = load_a_lds(cur, 32 * f, lane);
      kf1[f] = load_a_lds(cur + 16 * D_, 32 * f, lane);
    }
    v8f a0 = {}, a1 = {};
#pragma unroll
    for (int f = 0; f < 4; ++f) {
      a0 = wmma_f16(kf0[f], bQ[f], a0);
      a1 = wmma_f16(kf1[f], bQ[f], a1);
    }
    float vmax = -INFINITY;
#pragma unroll
    for (int j = 0; j < 8; ++j)
      vmax = fmaxf(vmax, fmaxf(a0[j] * SCALE_, a1[j] * SCALE_));
    vmax = fmaxf(vmax, __shfl_xor(vmax, 16, 32));
    const float nm  = fmaxf(mrun, vmax);
    const float fac = __expf(mrun - nm);
    float psum = 0.0f;
#pragma unroll
    for (int j = 0; j < 8; ++j)
      psum += __expf(a0[j] * SCALE_ - nm) + __expf(a1[j] * SCALE_ - nm);
    psum += __shfl_xor(psum, 16, 32);
    lrun = lrun * fac + psum;
    mrun = nm;
  }
  const float invl = 1.0f / lrun;

  float mS[8], ilS[8];
#pragma unroll
  for (int j = 0; j < 8; ++j) {
    mS[j]  = __shfl(mrun, j + shl, 32);
    ilS[j] = __shfl(invl, j + shl, 32);
  }

  // ---- Pass B ----
  v16h aQ[4];
#pragma unroll
  for (int f = 0; f < 4; ++f) aQ[f] = load_a_f32(qrow, 32 * f, lane);

  issue_ktile(Kh, hb, 0, sKoff[0], tid);
#pragma unroll 1
  for (int ct = 0; ct < N_ / 32; ++ct) {
    wait_async0();
    __syncthreads();
    if (ct + 1 < N_ / 32) issue_ktile(Kh, hb, ct + 1, sKoff[(ct + 1) & 1], tid);
    const _Float16* cur = sK + (ct & 1) * 32 * D_;

    v16h kb0[4], kb1[4];
#pragma unroll
    for (int f = 0; f < 4; ++f) {
      kb0[f] = load_b_lds(cur + lcol * D_ + 32 * f, lane);
      kb1[f] = load_b_lds(cur + (16 + lcol) * D_ + 32 * f, lane);
    }
    v8f a0 = {}, a1 = {};
#pragma unroll
    for (int f = 0; f < 4; ++f) {
      a0 = wmma_f16(aQ[f], kb0[f], a0);
      a1 = wmma_f16(aQ[f], kb1[f], a1);
    }
    float p0 = 0.0f, p1 = 0.0f;
#pragma unroll
    for (int j = 0; j < 8; ++j) {
      p0 += __expf(a0[j] * SCALE_ - mS[j]) * ilS[j];
      p1 += __expf(a1[j] * SCALE_ - mS[j]) * ilS[j];
    }
    atomicAdd(&sCs[ct * 32 + lcol], p0);
    atomicAdd(&sCs[ct * 32 + 16 + lcol], p1);
  }

  __syncthreads();
  for (int i = tid; i < N_; i += 384)
    CS[(size_t)blockIdx.x * N_ + i] = sCs[i];
}

// ---------------------------------------------------------------------------
// Kernel 2: per-(h,m) top-k selection, mask composition, and column
// compaction (index list + count) via block-wide prefix scan.
// ---------------------------------------------------------------------------
__global__ __launch_bounds__(256) void topk_mask_kernel(
    const float* __restrict__ CS, const int* __restrict__ LR,
    const int* __restrict__ TK, int* __restrict__ IDX, int* __restrict__ CNT) {
  const int g   = blockIdx.x;
  const int tid = threadIdx.x;
  __shared__ float sV[4096];
  __shared__ int   sI[4096];
  __shared__ unsigned char sT[N_];
  __shared__ int sOff[256];
  __shared__ int sCnt;

  const int tk = TK[0];

  for (int i = tid; i < 4096; i += 256) {
    sV[i] = (i < N_) ? CS[(size_t)g * N_ + i] : -INFINITY;
    sI[i] = i;
  }
  for (int i = tid; i < N_; i += 256) sT[i] = 0;
  if (tid == 0) sCnt = 0;

  for (int k = 2; k <= 4096; k <<= 1) {
    for (int j = k >> 1; j > 0; j >>= 1) {
      __syncthreads();
      for (int t = tid; t < 2048; t += 256) {
        const int i = ((t & ~(j - 1)) << 1) | (t & (j - 1));
        const int p = i | j;
        const bool dir = ((i & k) == 0);
        const float vi = sV[i], vp = sV[p];
        const int   ii = sI[i], ip = sI[p];
        const bool p_first = (vp > vi) || (vp == vi && ip < ii);
        const bool i_first = (vi > vp) || (vi == vp && ii < ip);
        if (dir ? p_first : i_first) {
          sV[i] = vp; sV[p] = vi; sI[i] = ip; sI[p] = ii;
        }
      }
    }
  }
  __syncthreads();

  for (int t = tid; t < tk && t < 4096; t += 256) {
    const int ix = sI[t];
    if (ix < N_) sT[ix] = 1;
  }
  int local = 0;
  for (int i = tid; i < N_; i += 256) local += (LR[(size_t)g * N_ + i] != 0);
  atomicAdd(&sCnt, local);
  __syncthreads();

  const bool gate = (sCnt + tk) < N_;

  const int base = tid * (N_ / 256);
  unsigned int mybits = 0;
  int mycnt = 0;
#pragma unroll
  for (int q = 0; q < N_ / 256; ++q) {
    const int i = base + q;
    const bool mk = ((sT[i] && gate) || (LR[(size_t)g * N_ + i] != 0));
    mybits |= (mk ? 1u : 0u) << q;
    mycnt += mk ? 1 : 0;
  }
  sOff[tid] = mycnt;
  __syncthreads();
  for (int d = 1; d < 256; d <<= 1) {
    const int add = (tid >= d) ? sOff[tid - d] : 0;
    __syncthreads();
    sOff[tid] += add;
    __syncthreads();
  }
  int pos = sOff[tid] - mycnt;
#pragma unroll
  for (int q = 0; q < N_ / 256; ++q) {
    if ((mybits >> q) & 1u) IDX[(size_t)g * N_ + pos++] = base + q;
  }
  if (tid == 0) CNT[g] = sOff[255];
}

// ---------------------------------------------------------------------------
// Kernel 3: gathered block-sparse flash attention per (h,m) group.
// Only the cnt masked columns are processed, in compacted 32-column tiles.
// K columns gathered via async-to-LDS (per-lane addresses); V gathered
// with f32 loads + f16 convert + transpose scatter (zero-filled tail).
// ---------------------------------------------------------------------------
__global__ __launch_bounds__(384) void csp_attn_kernel(
    const float* __restrict__ Q, const _Float16* __restrict__ Kh,
    const float* __restrict__ V, const int* __restrict__ IDX,
    const int* __restrict__ CNT, float* __restrict__ O) {
  const int hb   = blockIdx.x / M_;
  const int mb   = blockIdx.x % M_;
  const int tid  = threadIdx.x;
  const int wave = tid >> 5;
  const int lane = tid & 31;
  const int lcol = lane & 15;
  const int half = lane >> 4;
  const int shl  = half << 3;

  __shared__ __align__(16) _Float16 sK[32 * D_];    // gathered K tile [32][128]
  __shared__ __align__(16) _Float16 sVt[D_ * 32];   // gathered V tile [128][32]

  const unsigned sKoff = lds_off32(&sK[0]);

  const int cnt    = CNT[blockIdx.x];
  const int nTiles = (cnt + 31) >> 5;
  const int* idxp  = IDX + (size_t)blockIdx.x * N_;

  const int row0 = mb * QB_ + wave * 16;
  const float* qrow = Q + ((size_t)(hb * N_ + row0 + lcol)) * D_;
  v16h bQ[4];
#pragma unroll
  for (int f = 0; f < 4; ++f) bQ[f] = load_b_f32(qrow, 32 * f, lane);

  float mrun = -INFINITY, lrun = 0.0f;
  v8f o[8] = {};

#pragma unroll 1
  for (int it = 0; it < nTiles; ++it) {
    const int tileBase = it << 5;
    const int valid    = cnt - tileBase;

    __syncthreads();
    // K gather: async 16B chunks (stale tail slots are overridden below)
    for (int i = tid; i < 512; i += 384) {
      const int e = i << 3, slot = e >> 7, c = e & 127;
      const int pos = tileBase + slot;
      if (pos < cnt) {
        const int col = idxp[pos];
        async_ld16(Kh + ((size_t)(hb * N_ + col)) * D_ + c,
                   sKoff + (unsigned)((slot * D_ + c) * 2));
      }
    }
    // V gather + convert + transpose scatter (zero-fill tail: avoids 0*Inf)
    for (int i = tid; i < 1024; i += 384) {
      const int e = i << 2, slot = e >> 7, c = e & 127;
      float4 vv = {0.f, 0.f, 0.f, 0.f};
      const int pos = tileBase + slot;
      if (pos < cnt) {
        const int col = idxp[pos];
        vv = *(const float4*)&V[((size_t)(hb * N_ + col)) * D_ + c];
      }
      sVt[(c + 0) * 32 + slot] = (_Float16)vv.x;
      sVt[(c + 1) * 32 + slot] = (_Float16)vv.y;
      sVt[(c + 2) * 32 + slot] = (_Float16)vv.z;
      sVt[(c + 3) * 32 + slot] = (_Float16)vv.w;
    }
    wait_async0();
    __syncthreads();

    // S^T tiles: acc0 = slots 0..15, acc1 = slots 16..31 (rows = lanes)
    v16h kf0[4], kf1[4];
#pragma unroll
    for (int f = 0; f < 4; ++f) {
      kf0[f] = load_a_lds(sK, 32 * f, lane);
      kf1[f] = load_a_lds(sK + 16 * D_, 32 * f, lane);
    }
    v8f a0 = {}, a1 = {};
#pragma unroll
    for (int f = 0; f < 4; ++f) {
      a0 = wmma_f16(kf0[f], bQ[f], a0);
      a1 = wmma_f16(kf1[f], bQ[f], a1);
    }

    float s0[8], s1[8];
    float vmax = -INFINITY;
#pragma unroll
    for (int j = 0; j < 8; ++j) {
      s0[j] = (j + shl      < valid) ? a0[j] * SCALE_ : NEGBIG_;
      s1[j] = (16 + j + shl < valid) ? a1[j] * SCALE_ : NEGBIG_;
      vmax = fmaxf(vmax, fmaxf(s0[j], s1[j]));
    }
    vmax = fmaxf(vmax, __shfl_xor(vmax, 16, 32));
    const float nm  = fmaxf(mrun, vmax);
    const float fac = __expf(mrun - nm);

    v16h aP;
    float psum = 0.0f;
#pragma unroll
    for (int j = 0; j < 8; ++j) {
      const float e0 = __expf(s0[j] - nm);
      const float e1 = __expf(s1[j] - nm);
      psum += e0 + e1;
      aP[j]     = (_Float16)e0;   // A-layout: cols {0..7}/{8..15}
      aP[8 + j] = (_Float16)e1;   //           cols {16..23}/{24..31}
    }
    psum += __shfl_xor(psum, 16, 32);
    lrun = lrun * fac + psum;
    mrun = nm;

    float facS[8];
#pragma unroll
    for (int j = 0; j < 8; ++j) facS[j] = __shfl(fac, j + shl, 32);
#pragma unroll
    for (int t = 0; t < 8; ++t)
#pragma unroll
      for (int j = 0; j < 8; ++j) o[t][j] *= facS[j];

#pragma unroll
    for (int t = 0; t < 8; ++t) {
      v16h bV = load_b_lds(&sVt[(t * 16 + lcol) * 32], lane);
      o[t] = wmma_f16(aP, bV, o[t]);
    }
  }

  const float invl = 1.0f / lrun;
  float ilS[8];
#pragma unroll
  for (int j = 0; j < 8; ++j) ilS[j] = __shfl(invl, j + shl, 32);

#pragma unroll
  for (int t = 0; t < 8; ++t) {
#pragma unroll
    for (int j = 0; j < 8; ++j) {
      const int r = row0 + j + (half << 3);
      O[((size_t)(hb * N_ + r)) * D_ + t * 16 + lcol] = o[t][j] * ilS[j];
    }
  }
}

// ---------------------------------------------------------------------------
extern "C" void kernel_launch(void* const* d_in, const int* in_sizes, int n_in,
                              void* d_out, int out_size, void* d_ws, size_t ws_size,
                              hipStream_t stream) {
  const float* q  = (const float*)d_in[0];
  const float* k  = (const float*)d_in[1];
  const float* v  = (const float*)d_in[2];
  const int*   lr = (const int*)d_in[3];   // jax bool -> harness integer
  const int*   tk = (const int*)d_in[4];   // scalar topk
  float* out = (float*)d_out;

  const size_t hnd = (size_t)H_ * N_ * D_;
  const size_t hmn = (size_t)H_ * M_ * N_;
  _Float16* kh = (_Float16*)d_ws;                                    // 18.9 MB
  float* cs  = (float*)((char*)d_ws + hnd * 2);                      // 4.5 MB
  int*   idx = (int*)((char*)d_ws + hnd * 2 + hmn * 4);              // 4.5 MB
  int*   cnt = (int*)((char*)d_ws + hnd * 2 + hmn * 8);              // 1.5 KB

  convert_k_kernel<<<dim3((unsigned)(hnd / 4 / 256)), dim3(256), 0, stream>>>(k, kh);
  colsum_kernel<<<dim3(H_ * M_), dim3(384), 0, stream>>>(q, kh, cs);
  topk_mask_kernel<<<dim3(H_ * M_), dim3(256), 0, stream>>>(cs, lr, tk, idx, cnt);
  csp_attn_kernel<<<dim3(H_ * M_), dim3(384), 0, stream>>>(q, kh, v, idx, cnt, out);
}
